// Decoder_9045201125559
// MI455X (gfx1250) — compile-verified, matrix operand order checked by reading, not run on hardware
//
#include <hip/hip_runtime.h>
#include <hip/hip_bf16.h>

// ---------------- problem constants ----------------
#define VOCAB 32000
#define EDIM  512
#define HDIM  1024
#define BB    32
#define TT    128
#define MTOK  (BB * TT)     // 4096 tokens
#define G4    (4 * HDIM)    // 4096 gate columns

typedef __attribute__((ext_vector_type(16))) __bf16 v16bf;
typedef __attribute__((ext_vector_type(8)))  float  v8f;
typedef __attribute__((ext_vector_type(4)))  unsigned int u32x4;
typedef __attribute__((ext_vector_type(8)))  int i32x8;
typedef __attribute__((ext_vector_type(4)))  int i32x4;

union Frag {           // 16 bf16 = 32 bytes = two 16B loads
    uint4  u[2];
    v16bf  b;
};

__device__ __forceinline__ unsigned short f2bf(float f) {
    unsigned u = __float_as_uint(f);
    unsigned r = u + 0x7FFFu + ((u >> 16) & 1u);   // round-to-nearest-even
    return (unsigned short)(r >> 16);
}
__device__ __forceinline__ float sigmoidf_(float x) {
    return 1.0f / (1.0f + __expf(-x));
}

// ---------------- TDM: 2-D tile Global -> LDS (ISA ch. 8 descriptor) ------
// tensor: rows x d0 (d0 contiguous, row pitch = stride_d0 elements, 2B elems)
// tile  : tile_d1 rows x tile_d0 elements, packed row-major into LDS.
__device__ __forceinline__ void tdm_load_2d(unsigned lds_byte_off,
                                            const unsigned short* gptr,
                                            unsigned tensor_d0,
                                            unsigned tensor_d1,
                                            unsigned tile_d0,
                                            unsigned tile_d1,
                                            unsigned stride_d0) {
    unsigned long long ga = (unsigned long long)(size_t)gptr;
    u32x4 g0;
    g0[0] = 1u;                                          // count=1 (valid), user mode
    g0[1] = lds_byte_off;                                // lds_addr
    g0[2] = (unsigned)(ga & 0xFFFFFFFFu);                // global_addr[31:0]
    g0[3] = (unsigned)((ga >> 32) & 0x01FFFFFFu) | (2u << 30); // addr[56:32] | type=2
    i32x8 g1;
    g1[0] = (int)(1u << 16);                             // data_size=1 (2 bytes)
    g1[1] = (int)((tensor_d0 & 0xFFFFu) << 16);          // tensor_dim0[15:0] @ bits63:48
    g1[2] = (int)(((tensor_d0 >> 16) & 0xFFFFu) |        // tensor_dim0[31:16]
                  ((tensor_d1 & 0xFFFFu) << 16));        // tensor_dim1[15:0]
    g1[3] = (int)(((tensor_d1 >> 16) & 0xFFFFu) |        // tensor_dim1[31:16]
                  ((tile_d0 & 0xFFFFu) << 16));          // tile_dim0
    g1[4] = (int)(tile_d1 & 0xFFFFu);                    // tile_dim1, tile_dim2=0
    g1[5] = (int)stride_d0;                              // tensor_dim0_stride[31:0]
    g1[6] = 0;                                           // stride hi / dim1_stride
    g1[7] = 0;
    i32x4 z4 = {0, 0, 0, 0};
#if __has_include(<hip/amd_detail/amd_gfx1250_TDM.h>)
    i32x8 z8 = {0, 0, 0, 0, 0, 0, 0, 0};
    __builtin_amdgcn_tensor_load_to_lds(g0, g1, z4, z4, z8, 0);
#else
    __builtin_amdgcn_tensor_load_to_lds(g0, g1, z4, z4, 0);
#endif
}

__device__ __forceinline__ unsigned lds_off(const void* p) {
    // LDS aperture: low 32 bits of the flat address are the LDS byte offset
    return (unsigned)(size_t)p;
}

// ---------------- prep: f32 -> bf16 (vectorized x4) ----------------
__global__ void cvt_bf16_kernel(const float* __restrict__ src,
                                unsigned short* __restrict__ dst, int n4) {
    int i = blockIdx.x * blockDim.x + threadIdx.x;
    if (i >= n4) return;
    float4 v = ((const float4*)src)[i];
    ushort4 o;
    o.x = f2bf(v.x); o.y = f2bf(v.y); o.z = f2bf(v.z); o.w = f2bf(v.w);
    ((ushort4*)dst)[i] = o;
}

// ---------------- prep: xb[m][k] = bf16(emb[dst[b,t]][k]) ----------------
__global__ void gather_kernel(const int* __restrict__ dst,
                              const float* __restrict__ emb,
                              unsigned short* __restrict__ xb) {
    int m = blockIdx.x;                 // token index 0..4095
    int b = m >> 7, t = m & 127;
    int tok = dst[b * (TT + 1) + t];
    const float* row = emb + (size_t)tok * EDIM;
    unsigned short* orow = xb + (size_t)m * EDIM;
    int k = threadIdx.x;
    orow[k]       = f2bf(row[k]);
    orow[k + 256] = f2bf(row[k + 256]);
}

// ---------------- prep: base[b][g] = h0[b,:]·W_hh[g,:] + b_ih[g]+b_hh[g] --
__global__ void base_kernel(const float* __restrict__ h0,
                            const float* __restrict__ W_hh,
                            const float* __restrict__ b_ih,
                            const float* __restrict__ b_hh,
                            float* __restrict__ base) {
    int idx = blockIdx.x * blockDim.x + threadIdx.x;   // 32*4096
    if (idx >= BB * G4) return;
    int b = idx >> 12, g = idx & (G4 - 1);
    const float4* hp = (const float4*)(h0  + (size_t)b * HDIM);
    const float4* wp = (const float4*)(W_hh + (size_t)g * HDIM);
    float acc = 0.f;
    #pragma unroll 4
    for (int k = 0; k < HDIM / 4; ++k) {
        float4 hv = hp[k], wv = wp[k];
        acc += hv.x * wv.x + hv.y * wv.y + hv.z * wv.z + hv.w * wv.w;
    }
    base[idx] = acc + b_ih[g] + b_hh[g];
}

// ---------------- fragment loader (ISA 7.12.2 layouts) ----------------
// 16x32 bf16 A (or 32x16 B by symmetry): lane holds row/col = lane%16,
// half-wave selects K phase {0-7,16-23} vs {8-15,24-31}.
__device__ __forceinline__ v16bf load_frag_rowmajor(const unsigned short* p,
                                                    int ld, int row, int k0,
                                                    int lane) {
    int half = lane >> 4;
    const unsigned short* r = p + (size_t)row * ld + k0 + half * 8;
    Frag f;
    f.u[0] = *(const uint4*)(r);        // K = phase..phase+7
    f.u[1] = *(const uint4*)(r + 16);   // K = phase+16..phase+23
    return f.b;
}

// ---------------- gates GEMM + fused LSTM cell ----------------
__global__ void
__launch_bounds__(256)
lstm_gemm_kernel(const unsigned short* __restrict__ xb,
                 const unsigned short* __restrict__ wih,
                 const float* __restrict__ base,
                 const float* __restrict__ c0,
                 unsigned short* __restrict__ hbf) {
    int tid  = threadIdx.x;
    int lane = tid & 31;
    int wave = tid >> 5;
    int m0 = blockIdx.x * 16;                         // token tile
    int n0 = (blockIdx.y * 8 + wave) * 16;            // hidden-col tile
    int row = lane & 15;

    v8f acc_i = {}, acc_f = {}, acc_g = {}, acc_o = {};

    for (int k0 = 0; k0 < EDIM; k0 += 32) {
        v16bf a = load_frag_rowmajor(xb, EDIM, m0 + row, k0, lane);
        v16bf bi = load_frag_rowmajor(wih, EDIM, 0 * HDIM + n0 + row, k0, lane);
        v16bf bf_ = load_frag_rowmajor(wih, EDIM, 1 * HDIM + n0 + row, k0, lane);
        v16bf bg = load_frag_rowmajor(wih, EDIM, 2 * HDIM + n0 + row, k0, lane);
        v16bf bo = load_frag_rowmajor(wih, EDIM, 3 * HDIM + n0 + row, k0, lane);
        acc_i = __builtin_amdgcn_wmma_f32_16x16x32_bf16(false, a, false, bi,  (short)0, acc_i, false, false);
        acc_f = __builtin_amdgcn_wmma_f32_16x16x32_bf16(false, a, false, bf_, (short)0, acc_f, false, false);
        acc_g = __builtin_amdgcn_wmma_f32_16x16x32_bf16(false, a, false, bg,  (short)0, acc_g, false, false);
        acc_o = __builtin_amdgcn_wmma_f32_16x16x32_bf16(false, a, false, bo,  (short)0, acc_o, false, false);
    }

    int half = lane >> 4;
    int n = n0 + (lane & 15);
    #pragma unroll
    for (int j = 0; j < 8; ++j) {
        int m = m0 + j + half * 8;
        int b = m >> 7;
        const float* bb_ = base + (size_t)b * G4;
        float gi = sigmoidf_(acc_i[j] + bb_[n]);
        float gf = sigmoidf_(acc_f[j] + bb_[HDIM + n]);
        float gg = tanhf   (acc_g[j] + bb_[2 * HDIM + n]);
        float go = sigmoidf_(acc_o[j] + bb_[3 * HDIM + n]);
        float c  = gf * c0[(size_t)b * HDIM + n] + gi * gg;
        float h  = go * tanhf(c);
        hbf[(size_t)m * HDIM + n] = f2bf(h);
    }
}

// ---------------- logits GEMM: out = h(4096x1024) . fc_w^T + fc_b --------
// TDM double-buffered: per K-step(32) wave0 DMAs the next 256x32 B tile and
// 32x32 A tile into LDS while all 8 waves run WMMAs out of the current one.
// Block tile 32(M) x 256(N); wave tile 16 x 64 (one A frag, 4 B frags).
__global__ void
__launch_bounds__(256)
logits_gemm_kernel(const unsigned short* __restrict__ hbf,
                   const unsigned short* __restrict__ fcw,
                   const float* __restrict__ fcb,
                   float* __restrict__ out) {
    __shared__ unsigned short Bs[2][256 * 32];   // 2 x 16 KB
    __shared__ unsigned short As[2][32 * 32];    // 2 x  2 KB

    int tid  = threadIdx.x;
    int lane = tid & 31;
    int wave = tid >> 5;
    int mblk = blockIdx.x * 32;                  // block's 32 token rows
    int nblk = blockIdx.y * 256;                 // block's 256 vocab cols
    int mloc = (wave & 1) * 16;                  // wave's M half
    int nloc = (wave >> 1) * 64;                 // wave's 64-col strip
    int row  = lane & 15;

    v8f acc[4] = {v8f{}, v8f{}, v8f{}, v8f{}};

    // prologue: DMA K-slab 0 into buffer 0
    if (tid < 32) {
        tdm_load_2d(lds_off(&Bs[0][0]), fcw + (size_t)nblk * HDIM,
                    HDIM, VOCAB, 32, 256, HDIM);
        tdm_load_2d(lds_off(&As[0][0]), hbf + (size_t)mblk * HDIM,
                    HDIM, MTOK, 32, 32, HDIM);
        __builtin_amdgcn_s_wait_tensorcnt(0);
    }
    __syncthreads();

    int buf = 0;
    for (int k0 = 0; k0 < HDIM; k0 += 32) {
        int nxt = buf ^ 1;
        if (k0 + 32 < HDIM && tid < 32) {        // DMA next K-slab
            tdm_load_2d(lds_off(&Bs[nxt][0]), fcw + (size_t)nblk * HDIM + k0 + 32,
                        HDIM, VOCAB, 32, 256, HDIM);
            tdm_load_2d(lds_off(&As[nxt][0]), hbf + (size_t)mblk * HDIM + k0 + 32,
                        HDIM, MTOK, 32, 32, HDIM);
        }
        // compute current K-slab from LDS (ds_load_b128 fragments)
        v16bf a = load_frag_rowmajor(&As[buf][0], 32, mloc + row, 0, lane);
        #pragma unroll
        for (int nn = 0; nn < 4; ++nn) {
            v16bf bfr = load_frag_rowmajor(&Bs[buf][0], 32, nloc + nn * 16 + row, 0, lane);
            acc[nn] = __builtin_amdgcn_wmma_f32_16x16x32_bf16(
                false, a, false, bfr, (short)0, acc[nn], false, false);
        }
        if (tid < 32) __builtin_amdgcn_s_wait_tensorcnt(0);
        __syncthreads();
        buf = nxt;
    }

    int half = lane >> 4;
    #pragma unroll
    for (int nn = 0; nn < 4; ++nn) {
        int n = nblk + nloc + nn * 16 + (lane & 15);
        float bias = fcb[n];
        #pragma unroll
        for (int j = 0; j < 8; ++j) {
            int m = mblk + mloc + j + half * 8;
            out[(size_t)m * VOCAB + n] = acc[nn][j] + bias;
        }
    }
}

// ---------------- host ----------------
extern "C" void kernel_launch(void* const* d_in, const int* in_sizes, int n_in,
                              void* d_out, int out_size, void* d_ws, size_t ws_size,
                              hipStream_t stream) {
    const int*   dst  = (const int*)  d_in[0];
    const float* h0   = (const float*)d_in[1];
    const float* c0   = (const float*)d_in[2];
    const float* emb  = (const float*)d_in[3];
    const float* W_ih = (const float*)d_in[4];
    const float* W_hh = (const float*)d_in[5];
    const float* b_ih = (const float*)d_in[6];
    const float* b_hh = (const float*)d_in[7];
    const float* fc_w = (const float*)d_in[8];
    const float* fc_b = (const float*)d_in[9];
    float* out = (float*)d_out;

    // workspace layout (256B aligned chunks)
    char* ws = (char*)d_ws;
    unsigned short* wih_bf = (unsigned short*)ws;                       // 4096*512*2   = 4 MB
    unsigned short* fcw_bf = (unsigned short*)(ws + 4195328);           // 32000*1024*2 = 65.5 MB
    unsigned short* xb     = (unsigned short*)(ws + 4195328 + 65536000);// 4096*512*2
    float*          base   = (float*)(ws + 4195328 + 65536000 + 4194304);       // 32*4096*4
    unsigned short* hbf    = (unsigned short*)(ws + 4195328 + 65536000 + 4194304 + 524288);

    // prep
    {
        int n4 = (G4 * EDIM) / 4;   // W_ih
        cvt_bf16_kernel<<<(n4 + 255) / 256, 256, 0, stream>>>(W_ih, wih_bf, n4);
    }
    {
        int n4 = (VOCAB * HDIM) / 4; // fc_w
        cvt_bf16_kernel<<<(n4 + 255) / 256, 256, 0, stream>>>(fc_w, fcw_bf, n4);
    }
    gather_kernel<<<MTOK, 256, 0, stream>>>(dst, emb, xb);
    base_kernel<<<(BB * G4) / 256, 256, 0, stream>>>(h0, W_hh, b_ih, b_hh, base);

    // gates GEMM + fused LSTM cell -> h (bf16)
    dim3 g1(MTOK / 16, HDIM / 128);
    lstm_gemm_kernel<<<g1, 256, 0, stream>>>(xb, wih_bf, base, c0, hbf);

    // logits GEMM (dominant: 268 GFLOP, bf16 WMMA, TDM-fed)
    dim3 g2(MTOK / 32, VOCAB / 256);
    logits_gemm_kernel<<<g2, 256, 0, stream>>>(hbf, fcw_bf, fc_b, out);
}
